// MultiHeadAttention_8881992368520
// MI455X (gfx1250) — matrix operand order repaired and checked by reading, not compiled, not run on hardware
//
#include <hip/hip_runtime.h>
#include <hip/hip_bf16.h>
#include <math.h>

// ---------------------------------------------------------------------------
// MultiHeadAttention forward for MI455X (gfx1250, wave32, WMMA f16->f32)
// ---------------------------------------------------------------------------

typedef __attribute__((ext_vector_type(16))) _Float16 v16h;
typedef __attribute__((ext_vector_type(8)))  _Float16 v8h;
typedef __attribute__((ext_vector_type(8)))  float    v8f;

#define B_SZ 2
#define L_SZ 2048
#define D_SZ 1024
#define H_SZ 16
#define DK_SZ 64
#define MASK_FILL -1e9f
#define LN_EPS 1e-5f

static __device__ inline v8f vzero8() {
  v8f z;
#pragma unroll
  for (int i = 0; i < 8; ++i) z[i] = 0.0f;
  return z;
}

// A-fragment (16x32 f16) from row-major f16 source, leading dim lda.
// lane L: M = m0 + L%16 ; element e -> K = k0 + (L/16)*8 + (e/8)*16 + e%8
static __device__ inline v16h load_a_f16(const _Float16* __restrict__ a, int lda,
                                         int m0, int k0, int lane) {
  const int m  = m0 + (lane & 15);
  const int kb = k0 + ((lane >> 4) << 3);
  const _Float16* p = a + (size_t)m * lda + kb;
  union { v16h v; v8h h[2]; } u;
  u.h[0] = *(const v8h*)(p);
  u.h[1] = *(const v8h*)(p + 16);
  return u.v;
}

// Same A-fragment but converting from f32 source on the fly.
static __device__ inline v16h load_a_f32(const float* __restrict__ a, int lda,
                                         int m0, int k0, int lane) {
  const int m  = m0 + (lane & 15);
  const int kb = k0 + ((lane >> 4) << 3);
  const float* p = a + (size_t)m * lda + kb;
  v16h r;
#pragma unroll
  for (int i = 0; i < 8; ++i) {
    r[i]     = (_Float16)p[i];
    r[i + 8] = (_Float16)p[i + 16];
  }
  return r;
}

// B-fragment (32x16 f16) from a [N][K]-stored (B-transposed) f16 matrix.
// lane L: N = n0 + L%16 ; element e -> K = k0 + (L/16)*8 + (e/8)*16 + e%8
static __device__ inline v16h load_bt_f16(const _Float16* __restrict__ bt, int ldk,
                                          int n0, int k0, int lane) {
  const int n  = n0 + (lane & 15);
  const int kb = k0 + ((lane >> 4) << 3);
  const _Float16* p = bt + (size_t)n * ldk + kb;
  union { v16h v; v8h h[2]; } u;
  u.h[0] = *(const v8h*)(p);
  u.h[1] = *(const v8h*)(p + 16);
  return u.v;
}

static __device__ inline v8f wmma16(v16h a, v16h b, v8f c) {
  return __builtin_amdgcn_wmma_f32_16x16x32_f16(false, a, false, b, (short)0, c,
                                                false, false);
}

// ---------------------------------------------------------------------------
// 1) Weight transpose + f32->f16: W[K][N] (1024x1024 f32) -> Wt[N][K] f16
// ---------------------------------------------------------------------------
__global__ void __launch_bounds__(256)
transpose_w_kernel(const float* __restrict__ w, _Float16* __restrict__ wt) {
  __shared__ float tile[32][33];
  const int bx = blockIdx.x * 32, by = blockIdx.y * 32;
  const int tx = threadIdx.x & 31, ty = threadIdx.x >> 5;  // 32x8
#pragma unroll
  for (int i = 0; i < 32; i += 8)
    tile[ty + i][tx] = w[(size_t)(by + ty + i) * D_SZ + bx + tx];
  __syncthreads();
#pragma unroll
  for (int i = 0; i < 32; i += 8)
    wt[(size_t)(bx + ty + i) * D_SZ + by + tx] = (_Float16)tile[tx][ty + i];
}

// ---------------------------------------------------------------------------
// 2) Projection GEMM: Y[4096][1024] = X(f32) @ W  (Wt stored [N][K] f16)
//    MODE 0: Y f16 [M][1024] natural (Q, K)
//    MODE 1: Y f16 V-transposed [bh][dk][L] (V)
//    block 128 (4 waves), wave tile 32(M) x 64(N). grid = (16, 32)
// ---------------------------------------------------------------------------
template <int MODE>
__global__ void __launch_bounds__(128)
proj_gemm_kernel(const float* __restrict__ X, const _Float16* __restrict__ Wt,
                 _Float16* __restrict__ Y) {
  const int lane = threadIdx.x & 31;
  const int wave = threadIdx.x >> 5;
  const int m0 = blockIdx.y * 128 + wave * 32;
  const int n0 = blockIdx.x * 64;

  v8f acc[2][4];
#pragma unroll
  for (int mi = 0; mi < 2; ++mi)
#pragma unroll
    for (int j = 0; j < 4; ++j) acc[mi][j] = vzero8();

  for (int k0 = 0; k0 < D_SZ; k0 += 32) {
    v16h a0 = load_a_f32(X, D_SZ, m0, k0, lane);
    v16h a1 = load_a_f32(X, D_SZ, m0 + 16, k0, lane);
    if (k0 + 32 < D_SZ) {  // gfx1250 global_prefetch_b8 of next K-slab
      const int m = m0 + (lane & 15);
      __builtin_prefetch(X + (size_t)m * D_SZ + k0 + 32, 0, 3);
      __builtin_prefetch(X + (size_t)(m + 16) * D_SZ + k0 + 32, 0, 3);
    }
#pragma unroll
    for (int j = 0; j < 4; ++j) {
      v16h b = load_bt_f16(Wt, D_SZ, n0 + j * 16, k0, lane);
      acc[0][j] = wmma16(a0, b, acc[0][j]);
      acc[1][j] = wmma16(a1, b, acc[1][j]);
    }
  }

  const int nc = n0 + (lane & 15);
#pragma unroll
  for (int mi = 0; mi < 2; ++mi) {
    const int mr = m0 + mi * 16 + ((lane >> 4) << 3);
#pragma unroll
    for (int j = 0; j < 4; ++j) {
      const int n = nc + j * 16;
#pragma unroll
      for (int r = 0; r < 8; ++r) {
        const int m = mr + r;
        const _Float16 v = (_Float16)acc[mi][j][r];
        if (MODE == 0) {
          Y[(size_t)m * D_SZ + n] = v;
        } else {
          const int b = m >> 11, l = m & (L_SZ - 1);
          const int h = n >> 6, dk = n & (DK_SZ - 1);
          Y[(((size_t)(b * H_SZ + h) * DK_SZ + dk) << 11) + l] = v;
        }
      }
    }
  }
}

// ---------------------------------------------------------------------------
// 3) Scores: per (b,h)  S = Q_h @ K_h^T / 8, mask -> -1e9, f32 NT-stored.
//    K stored [L][D] already has the [N][K] shape needed for the B fragment.
//    grid = (L/64=32, L/128=16, B*H=32), block 128; wave tile 32(M) x 64(N).
// ---------------------------------------------------------------------------
__global__ void __launch_bounds__(128)
scores_kernel(const _Float16* __restrict__ Qf, const _Float16* __restrict__ Kf,
              const unsigned char* __restrict__ mask, float* __restrict__ attn) {
  const int lane = threadIdx.x & 31;
  const int wave = threadIdx.x >> 5;
  const int bh = blockIdx.z;
  const int b = bh >> 4, h = bh & 15;
  const int m0 = blockIdx.y * 128 + wave * 32;
  const int n0 = blockIdx.x * 64;

  const _Float16* Qb = Qf + (size_t)b * L_SZ * D_SZ + h * DK_SZ;
  const _Float16* Kb = Kf + (size_t)b * L_SZ * D_SZ + h * DK_SZ;

  v8f acc[2][4];
#pragma unroll
  for (int mi = 0; mi < 2; ++mi)
#pragma unroll
    for (int j = 0; j < 4; ++j) acc[mi][j] = vzero8();

#pragma unroll
  for (int k0 = 0; k0 < DK_SZ; k0 += 32) {
    v16h a0 = load_a_f16(Qb, D_SZ, m0, k0, lane);
    v16h a1 = load_a_f16(Qb, D_SZ, m0 + 16, k0, lane);
#pragma unroll
    for (int j = 0; j < 4; ++j) {
      v16h bf = load_bt_f16(Kb, D_SZ, n0 + j * 16, k0, lane);
      acc[0][j] = wmma16(a0, bf, acc[0][j]);
      acc[1][j] = wmma16(a1, bf, acc[1][j]);
    }
  }

  float* S = attn + (size_t)bh * L_SZ * L_SZ;
  const unsigned char* Mb = mask + (size_t)b * L_SZ * L_SZ;
  const int nc = n0 + (lane & 15);
  const float scale = 0.125f;  // 1/sqrt(64)
#pragma unroll
  for (int mi = 0; mi < 2; ++mi) {
    const int q0 = m0 + mi * 16 + ((lane >> 4) << 3);
#pragma unroll
    for (int j = 0; j < 4; ++j) {
      const int n = nc + j * 16;
#pragma unroll
      for (int r = 0; r < 8; ++r) {
        const int q = q0 + r;
        float s = acc[mi][j][r] * scale;
        if (Mb[(size_t)q * L_SZ + n]) s = MASK_FILL;
        __builtin_nontemporal_store(s, S + (size_t)q * L_SZ + n);
      }
    }
  }
}

// ---------------------------------------------------------------------------
// 4) Row softmax in place over 2048 f32. One 256-thread block per row.
// ---------------------------------------------------------------------------
__global__ void __launch_bounds__(256)
softmax_kernel(float* __restrict__ attn) {
  __shared__ float sm[256];
  const int tid = threadIdx.x;
  float* p = attn + (size_t)blockIdx.x * L_SZ;

  float vals[8];
  float lmax = -INFINITY;
#pragma unroll
  for (int i = 0; i < 8; ++i) {
    vals[i] = p[tid + i * 256];
    lmax = fmaxf(lmax, vals[i]);
  }
  sm[tid] = lmax;
  __syncthreads();
  for (int s = 128; s > 0; s >>= 1) {
    if (tid < s) sm[tid] = fmaxf(sm[tid], sm[tid + s]);
    __syncthreads();
  }
  const float rmax = sm[0];
  __syncthreads();

  float lsum = 0.0f;
#pragma unroll
  for (int i = 0; i < 8; ++i) {
    vals[i] = __expf(vals[i] - rmax);
    lsum += vals[i];
  }
  sm[tid] = lsum;
  __syncthreads();
  for (int s = 128; s > 0; s >>= 1) {
    if (tid < s) sm[tid] += sm[tid + s];
    __syncthreads();
  }
  const float inv = 1.0f / sm[0];
#pragma unroll
  for (int i = 0; i < 8; ++i) p[tid + i * 256] = vals[i] * inv;
}

// ---------------------------------------------------------------------------
// 5) Context: per (b,h)  C[2048][64] = attn(f32) @ V_h ; Vt stored [bh][dk][L].
//    grid = (L/128=16, B*H=32), block 128; wave tile 32(M) x 64(N=full dk).
// ---------------------------------------------------------------------------
__global__ void __launch_bounds__(128)
context_gemm_kernel(const float* __restrict__ attn, const _Float16* __restrict__ Vt,
                    _Float16* __restrict__ ctx) {
  const int lane = threadIdx.x & 31;
  const int wave = threadIdx.x >> 5;
  const int bh = blockIdx.y;
  const int b = bh >> 4, h = bh & 15;
  const int m0 = blockIdx.x * 128 + wave * 32;

  const float* A = attn + (size_t)bh * L_SZ * L_SZ;
  const _Float16* Bt = Vt + (size_t)bh * DK_SZ * L_SZ;

  v8f acc[2][4];
#pragma unroll
  for (int mi = 0; mi < 2; ++mi)
#pragma unroll
    for (int j = 0; j < 4; ++j) acc[mi][j] = vzero8();

  for (int k0 = 0; k0 < L_SZ; k0 += 32) {
    v16h a0 = load_a_f32(A, L_SZ, m0, k0, lane);
    v16h a1 = load_a_f32(A, L_SZ, m0 + 16, k0, lane);
    if (k0 + 32 < L_SZ) {  // prefetch next slab of attn rows
      const int m = m0 + (lane & 15);
      __builtin_prefetch(A + (size_t)m * L_SZ + k0 + 32, 0, 3);
      __builtin_prefetch(A + (size_t)(m + 16) * L_SZ + k0 + 32, 0, 3);
    }
#pragma unroll
    for (int j = 0; j < 4; ++j) {
      v16h bf = load_bt_f16(Bt, L_SZ, j * 16, k0, lane);
      acc[0][j] = wmma16(a0, bf, acc[0][j]);
      acc[1][j] = wmma16(a1, bf, acc[1][j]);
    }
  }

  const int nc = lane & 15;
#pragma unroll
  for (int mi = 0; mi < 2; ++mi) {
    const int q0 = m0 + mi * 16 + ((lane >> 4) << 3);
#pragma unroll
    for (int j = 0; j < 4; ++j) {
      const int dk = nc + j * 16;
#pragma unroll
      for (int r = 0; r < 8; ++r) {
        const int q = q0 + r;
        ctx[(size_t)(b * L_SZ + q) * D_SZ + h * DK_SZ + dk] = (_Float16)acc[mi][j][r];
      }
    }
  }
}

// ---------------------------------------------------------------------------
// 6) Output GEMM: out[4096][1024] = ctx(f16) @ W_fc + residual(input_Q), f32.
//    grid = (16, 32), block 128; wave tile 32(M) x 64(N).
// ---------------------------------------------------------------------------
__global__ void __launch_bounds__(128)
out_gemm_kernel(const _Float16* __restrict__ ctx, const _Float16* __restrict__ WtF,
                const float* __restrict__ resid, float* __restrict__ out) {
  const int lane = threadIdx.x & 31;
  const int wave = threadIdx.x >> 5;
  const int m0 = blockIdx.y * 128 + wave * 32;
  const int n0 = blockIdx.x * 64;

  v8f acc[2][4];
#pragma unroll
  for (int mi = 0; mi < 2; ++mi)
#pragma unroll
    for (int j = 0; j < 4; ++j) acc[mi][j] = vzero8();

  for (int k0 = 0; k0 < D_SZ; k0 += 32) {
    v16h a0 = load_a_f16(ctx, D_SZ, m0, k0, lane);
    v16h a1 = load_a_f16(ctx, D_SZ, m0 + 16, k0, lane);
#pragma unroll
    for (int j = 0; j < 4; ++j) {
      v16h b = load_bt_f16(WtF, D_SZ, n0 + j * 16, k0, lane);
      acc[0][j] = wmma16(a0, b, acc[0][j]);
      acc[1][j] = wmma16(a1, b, acc[1][j]);
    }
  }

  const int nc = n0 + (lane & 15);
#pragma unroll
  for (int mi = 0; mi < 2; ++mi) {
    const int mr = m0 + mi * 16 + ((lane >> 4) << 3);
#pragma unroll
    for (int j = 0; j < 4; ++j) {
      const int n = nc + j * 16;
#pragma unroll
      for (int r = 0; r < 8; ++r) {
        const int m = mr + r;
        out[(size_t)m * D_SZ + n] = acc[mi][j][r] + resid[(size_t)m * D_SZ + n];
      }
    }
  }
}

// ---------------------------------------------------------------------------
// 7) LayerNorm in place over 1024 f32 per row. One 256-thread block per row.
// ---------------------------------------------------------------------------
__global__ void __launch_bounds__(256)
layernorm_kernel(float* __restrict__ out, const float* __restrict__ gamma,
                 const float* __restrict__ beta) {
  __shared__ float sm[256];
  const int tid = threadIdx.x;
  float* p = out + (size_t)blockIdx.x * D_SZ;

  float v[4];
  float s = 0.0f;
#pragma unroll
  for (int i = 0; i < 4; ++i) {
    v[i] = p[tid + i * 256];
    s += v[i];
  }
  sm[tid] = s;
  __syncthreads();
  for (int st = 128; st > 0; st >>= 1) {
    if (tid < st) sm[tid] += sm[tid + st];
    __syncthreads();
  }
  const float mean = sm[0] * (1.0f / D_SZ);
  __syncthreads();

  float vs = 0.0f;
#pragma unroll
  for (int i = 0; i < 4; ++i) {
    const float d = v[i] - mean;
    vs += d * d;
  }
  sm[tid] = vs;
  __syncthreads();
  for (int st = 128; st > 0; st >>= 1) {
    if (tid < st) sm[tid] += sm[tid + st];
    __syncthreads();
  }
  const float rstd = rsqrtf(sm[0] * (1.0f / D_SZ) + LN_EPS);
#pragma unroll
  for (int i = 0; i < 4; ++i) {
    const int c = tid + i * 256;
    p[c] = (v[i] - mean) * rstd * gamma[c] + beta[c];
  }
}

// ---------------------------------------------------------------------------
extern "C" void kernel_launch(void* const* d_in, const int* in_sizes, int n_in,
                              void* d_out, int out_size, void* d_ws, size_t ws_size,
                              hipStream_t stream) {
  (void)in_sizes; (void)n_in; (void)out_size; (void)ws_size;

  const float* input_Q = (const float*)d_in[0];
  const float* input_K = (const float*)d_in[1];
  const float* input_V = (const float*)d_in[2];
  const unsigned char* attn_mask = (const unsigned char*)d_in[3];  // jnp.bool_ = 1 byte
  const float* W_Q  = (const float*)d_in[4];
  const float* W_K  = (const float*)d_in[5];
  const float* W_V  = (const float*)d_in[6];
  const float* W_fc = (const float*)d_in[7];
  const float* ln_gamma = (const float*)d_in[8];
  const float* ln_beta  = (const float*)d_in[9];

  float* out_ln = (float*)d_out;                         // [B,L,D]   = 4,194,304 f32
  float* attn   = out_ln + (size_t)B_SZ * L_SZ * D_SZ;   // [B,H,L,L] = 134,217,728 f32

  // Workspace (f16): Qf, Kf, Vt, ctx (4,194,304 halves each) + 4 weights
  _Float16* Qf  = (_Float16*)d_ws;
  _Float16* Kf  = Qf  + (size_t)4194304;
  _Float16* Vt  = Kf  + (size_t)4194304;
  _Float16* ctx = Vt  + (size_t)4194304;
  _Float16* WtQ = ctx + (size_t)4194304;
  _Float16* WtK = WtQ + (size_t)1048576;
  _Float16* WtV = WtK + (size_t)1048576;
  _Float16* WtF = WtV + (size_t)1048576;

  // 1) transpose weights to f16 [N][K]
  {
    dim3 g(32, 32), blk(256);
    transpose_w_kernel<<<g, blk, 0, stream>>>(W_Q,  WtQ);
    transpose_w_kernel<<<g, blk, 0, stream>>>(W_K,  WtK);
    transpose_w_kernel<<<g, blk, 0, stream>>>(W_V,  WtV);
    transpose_w_kernel<<<g, blk, 0, stream>>>(W_fc, WtF);
  }

  // 2) QKV projections
  {
    dim3 g(16, 32), blk(128);
    proj_gemm_kernel<0><<<g, blk, 0, stream>>>(input_Q, WtQ, Qf);
    proj_gemm_kernel<0><<<g, blk, 0, stream>>>(input_K, WtK, Kf);
    proj_gemm_kernel<1><<<g, blk, 0, stream>>>(input_V, WtV, Vt);
  }

  // 3) masked scores -> attn region (raw, pre-softmax)
  {
    dim3 g(32, 16, 32), blk(128);
    scores_kernel<<<g, blk, 0, stream>>>(Qf, Kf, attn_mask, attn);
  }

  // 4) softmax in place (B*H*L rows)
  {
    dim3 g(B_SZ * H_SZ * L_SZ), blk(256);
    softmax_kernel<<<g, blk, 0, stream>>>(attn);
  }

  // 5) context = attn @ V
  {
    dim3 g(16, 32), blk(128);
    context_gemm_kernel<<<g, blk, 0, stream>>>(attn, Vt, ctx);
  }

  // 6) output projection + residual
  {
    dim3 g(16, 32), blk(128);
    out_gemm_kernel<<<g, blk, 0, stream>>>(ctx, WtF, input_Q, out_ln);
  }

  // 7) layernorm in place
  {
    dim3 g(B_SZ * L_SZ), blk(256);
    layernorm_kernel<<<g, blk, 0, stream>>>(out_ln, ln_gamma, ln_beta);
  }
}
